// TransformerEncoder_78563541778634
// MI455X (gfx1250) — matrix operand order repaired and checked by reading, not compiled, not run on hardware
//
#include <hip/hip_runtime.h>
#include <hip/hip_bf16.h>
#include <math.h>
#include <stdint.h>

typedef __attribute__((ext_vector_type(16))) _Float16 v16h;
typedef __attribute__((ext_vector_type(8)))  _Float16 v8h;
typedef __attribute__((ext_vector_type(8)))  float    v8f;

#define NROWS 16384
#define DMODEL 256
#define IDIM_ 345
#define KPAD_IN 352
#define FF_ 1024
#define TSEQ 1024
#define NBATCH 16
#define NHEAD 4
#define DK_ 64
#define EPS_ 1e-5f

__device__ __forceinline__ v16h mk16(v8h lo, v8h hi) {
  v16h r;
#pragma unroll
  for (int j = 0; j < 8; ++j) { r[j] = lo[j]; r[j + 8] = hi[j]; }
  return r;
}

__device__ __forceinline__ v8f wmma_f16(v16h a, v16h b, v8f c) {
  return __builtin_amdgcn_wmma_f32_16x16x32_f16(
      false, a, false, b, (short)0, c, false, false);
}

// CDNA5 async copy: 16B per lane, global -> LDS, tracked by ASYNCcnt.
__device__ __forceinline__ void async_b128(uint32_t lds_off, uint64_t gaddr) {
  asm volatile("global_load_async_to_lds_b128 %0, %1, off"
               :: "v"(lds_off), "v"(gaddr) : "memory");
}
__device__ __forceinline__ void wait_async0() {
  asm volatile("s_wait_asynccnt 0x0" ::: "memory");
}
__device__ __forceinline__ void wait_async4() {   // allow 4 newest in flight
  asm volatile("s_wait_asynccnt 0x4" ::: "memory");
}

__device__ __forceinline__ float wave_sum(float v) {
#pragma unroll
  for (int off = 16; off > 0; off >>= 1) v += __shfl_xor(v, off, 32);
  return v;
}

// ---------------------------------------------------------------------------
// GEMM: C[M,N] = fuse(A[M,K](f16) @ BT[N,K](f16)^T + bias)
// FUSE: 0 none, 1 += resid(f32), 2 relu. OUTF16 selects f16 vs f32 output.
// 256 threads (8 waves), block tile 128x128, K-step 32, wave tile 32x64,
// double-buffered async-to-LDS staging, peeled pipeline. K % 32 == 0.
// ---------------------------------------------------------------------------
#define BM 128
#define BN 128
#define BK 32
#define TPITCH 40                    // halves per LDS row: 80B, 16B aligned
#define TILEB (BM * TPITCH * 2)      // bytes per LDS tile buffer

template<int FUSE, bool OUTF16, int N, int K>
__global__ __launch_bounds__(256)
void gemm_f16_kernel(const _Float16* __restrict__ A, const _Float16* __restrict__ BT,
                     const float* __restrict__ bias, const float* __restrict__ resid,
                     float* __restrict__ Cf, _Float16* __restrict__ Ch)
{
  __shared__ __align__(16) _Float16 As[2][BM * TPITCH];  // As[buf][m][k]
  __shared__ __align__(16) _Float16 Bs[2][BN * TPITCH];  // Bs[buf][n][k]

  const int tid = threadIdx.x;
  const int m0 = blockIdx.y * BM;
  const int n0 = blockIdx.x * BN;
  const int wid = tid >> 5, lane = tid & 31;
  const int wr = wid >> 1, wc = wid & 1;            // waves: 4(M) x 2(N)
  const int half = lane >> 4, l16 = lane & 15;

  const uint32_t asBase = (uint32_t)(uintptr_t)&As[0][0];
  const uint32_t bsBase = (uint32_t)(uintptr_t)&Bs[0][0];
  const int srow = tid >> 2, sch = tid & 3;

  auto stageAB = [&](int buf, int k0) {
#pragma unroll
    for (int j = 0; j < 2; ++j) {                   // 512 chunks / 256 thr = 2 each
      int row = srow + j * 64;
      uint32_t loff = (uint32_t)(row * (TPITCH * 2) + sch * 16);
      async_b128(asBase + buf * TILEB + loff,
                 (uint64_t)(uintptr_t)(A + (size_t)(m0 + row) * K + k0 + sch * 8));
      async_b128(bsBase + buf * TILEB + loff,
                 (uint64_t)(uintptr_t)(BT + (size_t)(n0 + row) * K + k0 + sch * 8));
    }
  };

  v8f acc[2][4] = {};

  auto computeK = [&](int cur) {
    const _Float16* Asc = As[cur];
    const _Float16* Bsc = Bs[cur];
    v16h afr[2], bfr[4];
#pragma unroll
    for (int tm = 0; tm < 2; ++tm) {
      const _Float16* ap = &Asc[(wr * 32 + tm * 16 + l16) * TPITCH];
      afr[tm] = mk16(*(const v8h*)(ap + half * 16), *(const v8h*)(ap + half * 16 + 8));
    }
#pragma unroll
    for (int tn = 0; tn < 4; ++tn) {
      const _Float16* bp = &Bsc[(wc * 64 + tn * 16 + l16) * TPITCH + half * 16];
      bfr[tn] = mk16(*(const v8h*)(bp), *(const v8h*)(bp + 8));
    }
#pragma unroll
    for (int tm = 0; tm < 2; ++tm)
#pragma unroll
      for (int tn = 0; tn < 4; ++tn)
        acc[tm][tn] = wmma_f16(afr[tm], bfr[tn], acc[tm][tn]);
  };

  stageAB(0, 0);
  wait_async0();
  __syncthreads();

  const int nk = K / BK;
#pragma unroll 1
  for (int kt = 0; kt < nk - 1; ++kt) {             // peeled: waits unconditional
    stageAB((kt + 1) & 1, (kt + 1) * BK);           // overlap with WMMA
    computeK(kt & 1);
    wait_async0();
    __syncthreads();
  }
  computeK((nk - 1) & 1);

  // straight-line epilogue
#pragma unroll
  for (int tm = 0; tm < 2; ++tm) {
#pragma unroll
    for (int tn = 0; tn < 4; ++tn) {
      const int gmb = m0 + wr * 32 + tm * 16 + 8 * half;
      const int gn  = n0 + wc * 64 + tn * 16 + l16;
      const float bb = bias[gn];
      float rv[8];
      if (FUSE == 1) {
#pragma unroll
        for (int r = 0; r < 8; ++r) rv[r] = resid[(size_t)(gmb + r) * N + gn];
      }
#pragma unroll
      for (int r = 0; r < 8; ++r) {
        float val = acc[tm][tn][r] + bb;
        if (FUSE == 1) val += rv[r];
        if (FUSE == 2) val = fmaxf(val, 0.f);
        if (OUTF16) Ch[(size_t)(gmb + r) * N + gn] = (_Float16)val;
        else        Cf[(size_t)(gmb + r) * N + gn] = val;
      }
    }
  }
}

// ---------------------------------------------------------------------------
// LayerNorm (D=256), one 256-thread block per row; wave reductions, 3 barriers.
// ---------------------------------------------------------------------------
__global__ __launch_bounds__(256)
void layernorm_kernel(const float* __restrict__ x, const float* __restrict__ s,
                      const float* __restrict__ bshift, float* __restrict__ y,
                      _Float16* __restrict__ y16)
{
  __shared__ float wred[8];
  const int row = blockIdx.x, tid = threadIdx.x;
  const int wid = tid >> 5, lane = tid & 31;
  const float v = x[(size_t)row * DMODEL + tid];

  const float s1 = wave_sum(v);
  if (lane == 0) wred[wid] = s1;
  __syncthreads();
  float tot = 0.f;
#pragma unroll
  for (int j = 0; j < 8; ++j) tot += wred[j];
  const float mean = tot * (1.f / DMODEL);
  const float d = v - mean;
  const float s2 = wave_sum(d * d);
  __syncthreads();
  if (lane == 0) wred[wid] = s2;
  __syncthreads();
  float vtot = 0.f;
#pragma unroll
  for (int j = 0; j < 8; ++j) vtot += wred[j];
  const float var = vtot * (1.f / DMODEL);
  const float o = d * rsqrtf(var + EPS_) * s[tid] + bshift[tid];
  y[(size_t)row * DMODEL + tid] = o;
  y16[(size_t)row * DMODEL + tid] = (_Float16)o;
}

// ---------------------------------------------------------------------------
// Attention: block per (b, h, 16-query tile); 128 threads = 4 waves.
// S padded to skew LDS banks; probabilities converted once to f16 (P16) so
// phase-3 A fragments are plain ds_load_b128 pairs.
// ---------------------------------------------------------------------------
#define QPITCHB 144                 // bytes per Qs/KT row (72 halves)
#define KTILEB  (16 * 72 * 2)       // bytes per K tile buffer
#define SPITCH  1032                // f32 words per S row (skews banks by 8)
#define PPITCH  1032                // halves per P16 row (2064B, 16B aligned)

__global__ __launch_bounds__(128)
void attn_kernel(const _Float16* __restrict__ q, const _Float16* __restrict__ k,
                 const _Float16* __restrict__ v, _Float16* __restrict__ o16)
{
  __shared__ float S[16 * SPITCH];                       // 66 KB scores
  __shared__ __align__(16) _Float16 P16[16 * PPITCH];    // 33 KB probs (f16)
  __shared__ __align__(16) _Float16 Qs[16 * 72];
  __shared__ __align__(16) _Float16 KT[4][2][16 * 72];   // per-wave dbl-buf K tiles
  __shared__ __align__(16) _Float16 VT[2][64 * 40];      // dbl-buf V^T [dk][kpos]
  __shared__ float red[16][8];

  const int b = blockIdx.z, h = blockIdx.y, q0 = blockIdx.x * 16;
  const int tid = threadIdx.x, wid = tid >> 5, lane = tid & 31;
  const int half = lane >> 4, l16 = lane & 15;
  const size_t rowbase = (size_t)b * TSEQ;
  const int colbase = h * DK_;

  // ---- stage Q (async): one 16B chunk per thread ----
  {
    const uint32_t qsBase = (uint32_t)(uintptr_t)Qs;
    int r = tid >> 3, ch = tid & 7;
    async_b128(qsBase + (uint32_t)(r * QPITCHB + ch * 16),
               (uint64_t)(uintptr_t)(q + (rowbase + q0 + r) * DMODEL + colbase + ch * 8));
    wait_async0();
  }
  __syncthreads();

  v16h aq[2];                                 // loop-invariant Q fragments
#pragma unroll
  for (int ks = 0; ks < 2; ++ks) {
    const _Float16* ap = &Qs[l16 * 72 + ks * 32];
    aq[ks] = mk16(*(const v8h*)(ap + half * 16), *(const v8h*)(ap + half * 16 + 8));
  }

  // ---- phase 1: S = 0.125 * Q K^T (peeled per-wave pipeline) ----
  const uint32_t ktBase = (uint32_t)(uintptr_t)&KT[wid][0][0];
  auto stageK = [&](int buf, int kb) {
#pragma unroll
    for (int j = 0; j < 4; ++j) {
      int i = lane + j * 32;
      int r = i >> 3, ch = i & 7;
      async_b128(ktBase + (uint32_t)(buf * KTILEB + r * QPITCHB + ch * 16),
                 (uint64_t)(uintptr_t)(k + (rowbase + kb + r) * DMODEL + colbase + ch * 8));
    }
  };
  auto p1comp = [&](int kb, const _Float16* kts) {
    v8f c8 = {};
#pragma unroll
    for (int ks = 0; ks < 2; ++ks) {
      const _Float16* bp = &kts[l16 * 72 + ks * 32 + half * 16];
      v16h bf = mk16(*(const v8h*)bp, *(const v8h*)(bp + 8));
      c8 = wmma_f16(aq[ks], bf, c8);
    }
#pragma unroll
    for (int r = 0; r < 8; ++r)
      S[(r + 8 * half) * SPITCH + kb + l16] = c8[r] * 0.125f;
  };

  stageK(0, wid * 16);
  int bufi = 0;
#pragma unroll 1
  for (int it = 0; it < 15; ++it) {               // uniform trip count -> no exec masks
    stageK(bufi ^ 1, (wid + (it + 1) * 4) * 16);
    wait_async4();                                // previous tile's 4 copies done
    p1comp((wid + it * 4) * 16, KT[wid][bufi]);
    bufi ^= 1;
  }
  wait_async0();
  p1comp((wid + 60) * 16, KT[wid][bufi]);
  __syncthreads();

  // ---- phase 2: softmax; write normalized probs to P16 (f16) ----
  {
    const int row = tid >> 3, sub = tid & 7;
    float mx = -3.4e38f;
    for (int c = sub; c < TSEQ; c += 8) mx = fmaxf(mx, S[row * SPITCH + c]);
    red[row][sub] = mx; __syncthreads();
    float rmax = red[row][0];
#pragma unroll
    for (int j = 1; j < 8; ++j) rmax = fmaxf(rmax, red[row][j]);
    __syncthreads();
    float sm = 0.f;
    for (int c = sub; c < TSEQ; c += 8) {
      float p = __expf(S[row * SPITCH + c] - rmax);
      S[row * SPITCH + c] = p;
      sm += p;
    }
    red[row][sub] = sm; __syncthreads();
    float rsum = 0.f;
#pragma unroll
    for (int j = 0; j < 8; ++j) rsum += red[row][j];
    const float inv = 1.f / rsum;
    for (int c = sub; c < TSEQ; c += 8)
      P16[row * PPITCH + c] = (_Float16)(S[row * SPITCH + c] * inv);
  }
  __syncthreads();

  // ---- phase 3: O = P V (double-buffered V tiles, peeled) ----
  auto stageV = [&](int buf, int kb) {            // 32 kpos x 64 dk -> VT[dk][kpos]
    const int kp2 = tid >> 3;                     // kpos pair index 0..15
    const int dc  = (tid & 7) * 8;                // dk chunk base
    const _Float16* g0 = v + (rowbase + kb + 2 * kp2) * DMODEL + colbase + dc;
    v8h a0 = *(const v8h*)g0;
    v8h a1 = *(const v8h*)(g0 + DMODEL);
#pragma unroll
    for (int u = 0; u < 8; ++u) {                 // pack 2 kpos -> one b32 store
      union { _Float16 hh[2]; uint32_t u32; } pk;
      pk.hh[0] = a0[u]; pk.hh[1] = a1[u];
      *(uint32_t*)&VT[buf][(dc + u) * 40 + 2 * kp2] = pk.u32;
    }
  };
  v8f oc = {};
  auto p3comp = [&](int kb, int cur) {
    const _Float16* ap = &P16[l16 * PPITCH + kb];
    v16h a = mk16(*(const v8h*)(ap + half * 16), *(const v8h*)(ap + half * 16 + 8));
    const _Float16* bp = &VT[cur][(wid * 16 + l16) * 40 + half * 16];
    v16h bf = mk16(*(const v8h*)bp, *(const v8h*)(bp + 8));
    oc = wmma_f16(a, bf, oc);
  };

  stageV(0, 0);
  __syncthreads();
#pragma unroll 1
  for (int kb = 0; kb < TSEQ - 32; kb += 32) {
    const int cur = (kb >> 5) & 1;
    stageV(cur ^ 1, kb + 32);
    p3comp(kb, cur);
    __syncthreads();
  }
  p3comp(TSEQ - 32, 1);

#pragma unroll
  for (int r = 0; r < 8; ++r)
    o16[(rowbase + q0 + r + 8 * half) * DMODEL + colbase + wid * 16 + l16] = (_Float16)oc[r];
}

// ---------------------------------------------------------------------------
// Conversion kernels (one-time per launch)
// ---------------------------------------------------------------------------
__global__ __launch_bounds__(256)
void pad_x_kernel(const float* __restrict__ x, _Float16* __restrict__ xp)
{
  const int m = blockIdx.x;
  for (int kk = threadIdx.x; kk < KPAD_IN; kk += 256)
    xp[(size_t)m * KPAD_IN + kk] =
        (kk < IDIM_) ? (_Float16)x[(size_t)m * IDIM_ + kk] : (_Float16)0.f;
}

// W[K,N] f32 -> WT[N,Kpad] f16 (transposed, K zero-padded). Grid: (ceil(Kpad/256), N)
__global__ __launch_bounds__(256)
void wtrans_kernel(const float* __restrict__ W, _Float16* __restrict__ WT,
                   int K, int N, int Kpad)
{
  const int kk = blockIdx.x * 256 + threadIdx.x;
  const int n  = blockIdx.y;
  if (kk >= Kpad) return;
  WT[(size_t)n * Kpad + kk] = (kk < K) ? (_Float16)W[(size_t)kk * N + n] : (_Float16)0.f;
}

// ---------------------------------------------------------------------------
extern "C" void kernel_launch(void* const* d_in, const int* in_sizes, int n_in,
                              void* d_out, int out_size, void* d_ws, size_t ws_size,
                              hipStream_t stream)
{
  (void)in_sizes; (void)n_in; (void)out_size; (void)ws_size;
  const float* x    = (const float*)d_in[0];
  const float* Win  = (const float*)d_in[2];
  const float* b_in = (const float*)d_in[3];
  const float* ln1s = (const float*)d_in[4];
  const float* ln1b = (const float*)d_in[5];
  const float* Wq   = (const float*)d_in[6];
  const float* bq   = (const float*)d_in[7];
  const float* Wk   = (const float*)d_in[8];
  const float* bk   = (const float*)d_in[9];
  const float* Wv   = (const float*)d_in[10];
  const float* bv   = (const float*)d_in[11];
  const float* Wo   = (const float*)d_in[12];
  const float* bo   = (const float*)d_in[13];
  const float* ln2s = (const float*)d_in[14];
  const float* ln2b = (const float*)d_in[15];
  const float* W1   = (const float*)d_in[16];
  const float* b1   = (const float*)d_in[17];
  const float* W2   = (const float*)d_in[18];
  const float* b2   = (const float*)d_in[19];
  const float* lnos = (const float*)d_in[20];
  const float* lnob = (const float*)d_in[21];
  float* out = (float*)d_out;

  const size_t ED = (size_t)NROWS * DMODEL;
  char* p = (char*)d_ws;
  auto carve = [&](size_t bytes) {
    char* r = p; p += (bytes + 255) & ~(size_t)255; return r;
  };
  float*    e    = (float*)   carve(ED * 4);
  _Float16* e16  = (_Float16*)carve(ED * 2);
  _Float16* q16  = (_Float16*)carve(ED * 2);
  _Float16* k16  = (_Float16*)carve(ED * 2);
  _Float16* v16  = (_Float16*)carve(ED * 2);
  _Float16* o16  = (_Float16*)carve(ED * 2);
  _Float16* h16  = (_Float16*)carve((size_t)NROWS * FF_ * 2);
  _Float16* xp16 = (_Float16*)carve((size_t)NROWS * KPAD_IN * 2);
  _Float16* WinT = (_Float16*)carve((size_t)DMODEL * KPAD_IN * 2);
  _Float16* WqT  = (_Float16*)carve((size_t)4 * DMODEL * DMODEL * 2);
  _Float16* WkT  = (_Float16*)carve((size_t)4 * DMODEL * DMODEL * 2);
  _Float16* WvT  = (_Float16*)carve((size_t)4 * DMODEL * DMODEL * 2);
  _Float16* WoT  = (_Float16*)carve((size_t)4 * DMODEL * DMODEL * 2);
  _Float16* W1T  = (_Float16*)carve((size_t)4 * FF_ * DMODEL * 2);
  _Float16* W2T  = (_Float16*)carve((size_t)4 * DMODEL * FF_ * 2);

  auto wtrans = [&](const float* W, _Float16* WT, int K, int N, int Kpad) {
    wtrans_kernel<<<dim3((unsigned)((Kpad + 255) / 256), (unsigned)N), dim3(256), 0, stream>>>(
        W, WT, K, N, Kpad);
  };

  // one-time conversions
  pad_x_kernel<<<NROWS, 256, 0, stream>>>(x, xp16);
  wtrans(Win, WinT, IDIM_, DMODEL, KPAD_IN);
  for (int i = 0; i < 4; ++i) {
    const size_t wdd = (size_t)i * DMODEL * DMODEL;
    const size_t wdf = (size_t)i * DMODEL * FF_;
    wtrans(Wq + wdd, WqT + wdd, DMODEL, DMODEL, DMODEL);
    wtrans(Wk + wdd, WkT + wdd, DMODEL, DMODEL, DMODEL);
    wtrans(Wv + wdd, WvT + wdd, DMODEL, DMODEL, DMODEL);
    wtrans(Wo + wdd, WoT + wdd, DMODEL, DMODEL, DMODEL);
    wtrans(W1 + wdf, W1T + wdf, DMODEL, FF_, DMODEL);   // -> [FF, D]
    wtrans(W2 + wdf, W2T + wdf, FF_, DMODEL, FF_);      // -> [D, FF]
  }

  const dim3 gN256(DMODEL / BN, NROWS / BM);   // (2, 128)
  const dim3 gN1024(FF_ / BN, NROWS / BM);     // (8, 128)
  const dim3 blk(256);

  // input projection: e = xpad @ Win + b_in
  gemm_f16_kernel<0, false, DMODEL, KPAD_IN><<<gN256, blk, 0, stream>>>(
      xp16, WinT, b_in, nullptr, e, nullptr);

  for (int i = 0; i < 4; ++i) {
    const size_t wdd = (size_t)i * DMODEL * DMODEL;
    const size_t wdf = (size_t)i * DMODEL * FF_;
    layernorm_kernel<<<NROWS, 256, 0, stream>>>(e, ln1s + i * DMODEL, ln1b + i * DMODEL, e, e16);
    gemm_f16_kernel<0, true, DMODEL, DMODEL><<<gN256, blk, 0, stream>>>(
        e16, WqT + wdd, bq + i * DMODEL, nullptr, nullptr, q16);
    gemm_f16_kernel<0, true, DMODEL, DMODEL><<<gN256, blk, 0, stream>>>(
        e16, WkT + wdd, bk + i * DMODEL, nullptr, nullptr, k16);
    gemm_f16_kernel<0, true, DMODEL, DMODEL><<<gN256, blk, 0, stream>>>(
        e16, WvT + wdd, bv + i * DMODEL, nullptr, nullptr, v16);
    attn_kernel<<<dim3(TSEQ / 16, NHEAD, NBATCH), dim3(128), 0, stream>>>(q16, k16, v16, o16);
    gemm_f16_kernel<1, false, DMODEL, DMODEL><<<gN256, blk, 0, stream>>>(
        o16, WoT + wdd, bo + i * DMODEL, e, e, nullptr);
    layernorm_kernel<<<NROWS, 256, 0, stream>>>(e, ln2s + i * DMODEL, ln2b + i * DMODEL, e, e16);
    gemm_f16_kernel<2, true, FF_, DMODEL><<<gN1024, blk, 0, stream>>>(
        e16, W1T + wdf, b1 + i * FF_, nullptr, nullptr, h16);
    gemm_f16_kernel<1, false, DMODEL, FF_><<<gN256, blk, 0, stream>>>(
        h16, W2T + wdf, b2 + i * DMODEL, e, e, nullptr);
  }
  layernorm_kernel<<<NROWS, 256, 0, stream>>>(e, lnos, lnob, out, e16);
}
